// _IPEXGatedMLPMOECPU_45956150067253
// MI455X (gfx1250) — matrix operand order repaired and checked by reading, not compiled, not run on hardware
//
#include <hip/hip_runtime.h>
#include <hip/hip_bf16.h>

// Problem constants (match reference)
#define T_TOK 2048
#define H_DIM 1024
#define I_DIM 2816
#define N_EXP 8
#define TOPK  2
#define ROWS  (T_TOK * TOPK)   // 4096 routed rows

#define KSLAB    128           // K-dim slab staged in LDS
#define SLDS     136           // padded LDS row stride (bf16 elems): 136*2B=272B, 68 dwords, 68%64=4
#define NB       256           // N columns covered per block (8 waves * 2 tiles * 16)

typedef __attribute__((ext_vector_type(16))) __bf16 v16bf;
typedef __attribute__((ext_vector_type(8)))  __bf16 v8bf;
typedef __attribute__((ext_vector_type(8)))  float  v8f;
typedef __attribute__((ext_vector_type(4)))  int    v4i;

typedef __attribute__((address_space(1))) v4i  v4i_g;   // global
typedef __attribute__((address_space(3))) v4i  v4i_l;   // LDS
typedef __attribute__((address_space(3))) void lds_void_t;

// Merge two v8bf fragments (A halves 0..7 / 8..15) into a v16bf WMMA operand.
__device__ __forceinline__ v16bf cat16(v8bf lo, v8bf hi) {
    return __builtin_shufflevector(lo, hi, 0, 1, 2, 3, 4, 5, 6, 7,
                                           8, 9, 10, 11, 12, 13, 14, 15);
}

// CDNA5 async global->LDS copy of 16 bytes per lane (ASYNCcnt-tracked).
__device__ __forceinline__ void async_copy_b128(const void* gsrc, void* lds_dst) {
#if __has_builtin(__builtin_amdgcn_global_load_async_to_lds_b128)
    __builtin_amdgcn_global_load_async_to_lds_b128(
        (v4i_g*)(unsigned long long)gsrc,   // numeric reinterpret: global == flat address
        (v4i_l*)lds_dst,                    // addrspacecast: VGPR holds LDS byte offset
        0, 0);
#else
    lds_void_t* lp = (lds_void_t*)lds_dst;
    asm volatile("global_load_async_to_lds_b128 %0, %1, off"
                 :: "v"(lp), "v"(gsrc)
                 : "memory");
#endif
}

__device__ __forceinline__ void wait_asynccnt0() {
#if __has_builtin(__builtin_amdgcn_s_wait_asynccnt)
    __builtin_amdgcn_s_wait_asynccnt(0);
#else
    asm volatile("s_wait_asynccnt 0x0" ::: "memory");
#endif
}

// ---------------------------------------------------------------------------
// Kernel 0: zero d_out, expert counts, bucket arrays
// ---------------------------------------------------------------------------
__global__ void k_zero(float* __restrict__ out, int* __restrict__ cnt,
                       int* __restrict__ rowTok, float* __restrict__ rowW) {
    int i = blockIdx.x * blockDim.x + threadIdx.x;
    if (i < T_TOK * H_DIM) out[i] = 0.0f;
    if (i < ROWS) { rowTok[i] = 0; rowW[i] = 0.0f; }
    if (i < N_EXP) cnt[i] = 0;
}

// ---------------------------------------------------------------------------
// Kernel 1: routing — softmax over 8 logits, top-2, renormalize.
// (softmax denominator cancels in the top-k renormalization)
// ---------------------------------------------------------------------------
__global__ void k_route(const float* __restrict__ logits, int* __restrict__ cnt,
                        int* __restrict__ topkId, float* __restrict__ topkW) {
    int t = blockIdx.x * blockDim.x + threadIdx.x;
    if (t >= T_TOK) return;
    float l[N_EXP];
    float mx = -1e30f;
    #pragma unroll
    for (int e = 0; e < N_EXP; ++e) { l[e] = logits[t * N_EXP + e]; mx = fmaxf(mx, l[e]); }
    #pragma unroll
    for (int e = 0; e < N_EXP; ++e) l[e] = __expf(l[e] - mx);
    int i0 = 0;
    #pragma unroll
    for (int e = 1; e < N_EXP; ++e) if (l[e] > l[i0]) i0 = e;
    int i1 = (i0 == 0) ? 1 : 0;
    #pragma unroll
    for (int e = 0; e < N_EXP; ++e) if (e != i0 && l[e] > l[i1]) i1 = e;
    float w0 = l[i0], w1 = l[i1];
    float inv = __builtin_amdgcn_rcpf(w0 + w1);
    topkId[t * 2 + 0] = i0;  topkW[t * 2 + 0] = w0 * inv;
    topkId[t * 2 + 1] = i1;  topkW[t * 2 + 1] = w1 * inv;
    atomicAdd(&cnt[i0], 1);
    atomicAdd(&cnt[i1], 1);
}

// ---------------------------------------------------------------------------
// Kernel 2: exclusive scan over 8 expert counts (single wave)
// ---------------------------------------------------------------------------
__global__ void k_scan(const int* __restrict__ cnt, int* __restrict__ off,
                       int* __restrict__ cursor) {
    if (threadIdx.x == 0) {
        int acc = 0;
        for (int e = 0; e < N_EXP; ++e) { off[e] = acc; cursor[e] = acc; acc += cnt[e]; }
    }
}

// ---------------------------------------------------------------------------
// Kernel 3: scatter tokens into per-expert buckets
// ---------------------------------------------------------------------------
__global__ void k_scatter(const int* __restrict__ topkId, const float* __restrict__ topkW,
                          int* __restrict__ cursor, int* __restrict__ rowTok,
                          float* __restrict__ rowW) {
    int t = blockIdx.x * blockDim.x + threadIdx.x;
    if (t >= T_TOK) return;
    #pragma unroll
    for (int j = 0; j < TOPK; ++j) {
        int e = topkId[t * 2 + j];
        int p = atomicAdd(&cursor[e], 1);
        rowTok[p] = t;
        rowW[p]   = topkW[t * 2 + j];
    }
}

// ---------------------------------------------------------------------------
// Fragment loads.
// A (16x32 bf16) from LDS slab: lane l -> row (l&15); half h ->
// K=kk+g*16+h, g = l>>4.  Two aligned 16B LDS reads.
// B (32x16 bf16) from global fp32 row: lane l -> column (l&15); half h ->
// K = kk+g*16+h, i.e. 16 contiguous floats converted to bf16.
// ---------------------------------------------------------------------------
__device__ __forceinline__ v16bf load_A_lds(const __bf16* sbase, int kk, int m, int g) {
    const v8bf lo = *(const v8bf*)(sbase + m * SLDS + kk + g * 16);
    const v8bf hi = *(const v8bf*)(sbase + m * SLDS + kk + g * 16 + 8);
    return cat16(lo, hi);
}

__device__ __forceinline__ v16bf load_B_f32(const float* row, int kk) {
    v16bf b;
    #pragma unroll
    for (int h = 0; h < 16; ++h) b[h] = (__bf16)row[kk + h];
    return b;
}

// ---------------------------------------------------------------------------
// GEMM1: h = silu(X W1^T) * (X W3^T), bf16 WMMA, fp32 accumulate.
// grid = (I/256, T/16, E), block = 256 (8 waves).
// Wave w computes output tiles [16 x 16] at columns n0 + {0,16}, n0 = bx*256 + w*32.
// X slab (16 rows x 128 K, fp32 -> bf16) staged in LDS per 128-K step.
// ---------------------------------------------------------------------------
__global__ __launch_bounds__(256)
void k_gemm1(const float* __restrict__ X, const float* __restrict__ W1,
             const float* __restrict__ W3, const int* __restrict__ rowTok,
             const int* __restrict__ off, const int* __restrict__ cnt,
             __bf16* __restrict__ hbuf) {
    __shared__ __bf16 sx[16 * SLDS];

    const int e  = blockIdx.z;
    const int ne = cnt[e];
    const int m0 = blockIdx.y * 16;
    if (m0 >= ne) return;

    const int wave = threadIdx.x >> 5;
    const int lane = threadIdx.x & 31;
    const int n    = lane & 15;
    const int g    = lane >> 4;
    const int n0   = blockIdx.x * NB + wave * 32;    // I-dim tile base (2 tiles)
    const int row0 = off[e] + m0;

    // Slab-loader mapping: 512 float4 per slab, 2 per thread.
    const int lrow0 = threadIdx.x >> 5;              // rows 0..7
    const int lcol0 = (threadIdx.x & 31) * 4;        // float index within slab
    const int lrow1 = lrow0 + 8;                     // rows 8..15
    const int tokA  = rowTok[row0 + lrow0];
    const int tokB  = rowTok[row0 + lrow1];
    const float* xA = X + (long)tokA * H_DIM + lcol0;
    const float* xB = X + (long)tokB * H_DIM + lcol0;

    const float* w1r0 = W1 + (long)e * I_DIM * H_DIM + (long)(n0 + n) * H_DIM + g * 16;
    const float* w1r1 = w1r0 + 16 * H_DIM;
    const float* w3r0 = W3 + (long)e * I_DIM * H_DIM + (long)(n0 + n) * H_DIM + g * 16;
    const float* w3r1 = w3r0 + 16 * H_DIM;

    v8f accg0 = {}, accg1 = {}, accu0 = {}, accu1 = {};

    for (int ks = 0; ks < H_DIM; ks += KSLAB) {
        __syncthreads();                             // WAR on sx
        {   // stage 16x128 fp32 -> bf16 into LDS
            float4 va = *(const float4*)(xA + ks);
            float4 vb = *(const float4*)(xB + ks);
            __bf16* da = &sx[lrow0 * SLDS + lcol0];
            __bf16* db = &sx[lrow1 * SLDS + lcol0];
            da[0] = (__bf16)va.x; da[1] = (__bf16)va.y; da[2] = (__bf16)va.z; da[3] = (__bf16)va.w;
            db[0] = (__bf16)vb.x; db[1] = (__bf16)vb.y; db[2] = (__bf16)vb.z; db[3] = (__bf16)vb.w;
        }
        __syncthreads();                             // RAW on sx

        if (ks + KSLAB < H_DIM) {                    // prefetch next weight slab
            __builtin_prefetch(w1r0 + ks + KSLAB, 0, 1);
            __builtin_prefetch(w3r0 + ks + KSLAB, 0, 1);
        }

        #pragma unroll
        for (int kk = 0; kk < KSLAB; kk += 32) {
            const v16bf a   = load_A_lds(sx, kk, n, g);
            const v16bf b10 = load_B_f32(w1r0 + ks, kk);
            const v16bf b11 = load_B_f32(w1r1 + ks, kk);
            const v16bf b30 = load_B_f32(w3r0 + ks, kk);
            const v16bf b31 = load_B_f32(w3r1 + ks, kk);
            accg0 = __builtin_amdgcn_wmma_f32_16x16x32_bf16(false, a, false, b10, (short)0, accg0, false, false);
            accg1 = __builtin_amdgcn_wmma_f32_16x16x32_bf16(false, a, false, b11, (short)0, accg1, false, false);
            accu0 = __builtin_amdgcn_wmma_f32_16x16x32_bf16(false, a, false, b30, (short)0, accu0, false, false);
            accu1 = __builtin_amdgcn_wmma_f32_16x16x32_bf16(false, a, false, b31, (short)0, accu1, false, false);
        }
    }

    // SwiGLU + store bf16 h tiles. Element (m, n): m = r + 8*g for VGPR r.
    #pragma unroll
    for (int r = 0; r < 8; ++r) {
        const int mm = r + 8 * g;
        __bf16* hrow = hbuf + (long)(row0 + mm) * I_DIM;
        float g0 = accg0[r], u0 = accu0[r];
        float g1 = accg1[r], u1 = accu1[r];
        float h0 = g0 * __builtin_amdgcn_rcpf(1.0f + __expf(-g0)) * u0;
        float h1 = g1 * __builtin_amdgcn_rcpf(1.0f + __expf(-g1)) * u1;
        hrow[n0 + n]      = (__bf16)h0;
        hrow[n0 + 16 + n] = (__bf16)h1;
    }
}

// ---------------------------------------------------------------------------
// GEMM2: y = h W2^T scaled by gate weight, atomic scatter-add into out.
// grid = (H/256, T/16, E), block = 256 (8 waves), 2 N-tiles per wave.
// h slab (16 rows x 128 K, bf16) staged in LDS via ASYNC global->LDS copies:
// each of the 256 threads moves one aligned 16B chunk (GLOBAL_LOAD_ASYNC_TO_
// LDS_B128), then waits ASYNCcnt and barriers before fragment reads.
// ---------------------------------------------------------------------------
__global__ __launch_bounds__(256)
void k_gemm2(const __bf16* __restrict__ hbuf, const float* __restrict__ W2,
             const int* __restrict__ rowTok, const float* __restrict__ rowW,
             const int* __restrict__ off, const int* __restrict__ cnt,
             float* __restrict__ out) {
    __shared__ __bf16 sh[16 * SLDS];

    const int e  = blockIdx.z;
    const int ne = cnt[e];
    const int m0 = blockIdx.y * 16;
    if (m0 >= ne) return;

    const int wave = threadIdx.x >> 5;
    const int lane = threadIdx.x & 31;
    const int n    = lane & 15;
    const int g    = lane >> 4;
    const int n0   = blockIdx.x * NB + wave * 32;    // H-dim tile base (2 tiles)
    const int row0 = off[e] + m0;

    // Slab-loader mapping: 16 rows x 16 chunks of 8 bf16 -> one b128 per thread.
    const int lrow = threadIdx.x >> 4;
    const int lc8  = (threadIdx.x & 15) * 8;
    const __bf16* hsrc = hbuf + (long)(row0 + lrow) * I_DIM + lc8;
    __bf16* sdst = &sh[lrow * SLDS + lc8];

    const float* b2r0 = W2 + (long)e * H_DIM * I_DIM + (long)(n0 + n) * I_DIM + g * 16;
    const float* b2r1 = b2r0 + 16 * I_DIM;

    v8f acc0 = {}, acc1 = {};

    for (int ks = 0; ks < I_DIM; ks += KSLAB) {      // 22 slabs
        __syncthreads();                             // WAR on sh
        async_copy_b128(hsrc + ks, sdst);            // ASYNCcnt-tracked DMA to LDS

        if (ks + KSLAB < I_DIM) {
            __builtin_prefetch(b2r0 + ks + KSLAB, 0, 1);
        }

        wait_asynccnt0();
        __syncthreads();                             // RAW on sh (all waves' copies)

        #pragma unroll
        for (int kk = 0; kk < KSLAB; kk += 32) {
            const v16bf a  = load_A_lds(sh, kk, n, g);
            const v16bf b0 = load_B_f32(b2r0 + ks, kk);
            const v16bf b1 = load_B_f32(b2r1 + ks, kk);
            acc0 = __builtin_amdgcn_wmma_f32_16x16x32_bf16(false, a, false, b0, (short)0, acc0, false, false);
            acc1 = __builtin_amdgcn_wmma_f32_16x16x32_bf16(false, a, false, b1, (short)0, acc1, false, false);
        }
    }

    #pragma unroll
    for (int r = 0; r < 8; ++r) {
        const int mm = r + 8 * g;
        const int t  = rowTok[row0 + mm];
        const float w = rowW[row0 + mm];
        float* orow = out + (long)t * H_DIM;
        atomicAdd(&orow[n0 + n],      w * acc0[r]);
        atomicAdd(&orow[n0 + 16 + n], w * acc1[r]);
    }
}

// ---------------------------------------------------------------------------
// Launcher
// ---------------------------------------------------------------------------
extern "C" void kernel_launch(void* const* d_in, const int* in_sizes, int n_in,
                              void* d_out, int out_size, void* d_ws, size_t ws_size,
                              hipStream_t stream) {
    const float* X      = (const float*)d_in[0];   // [T, H]
    const float* logits = (const float*)d_in[1];   // [T, E]
    const float* W1     = (const float*)d_in[2];   // [E, I, H]
    const float* W3     = (const float*)d_in[3];   // [E, I, H]
    const float* W2     = (const float*)d_in[4];   // [E, H, I]
    float* out = (float*)d_out;                    // [T, H]

    // Workspace carve-up
    int*    cnt    = (int*)d_ws;                   // [E]
    int*    off    = cnt + N_EXP;                  // [E]
    int*    cursor = off + N_EXP;                  // [E]
    int*    topkId = cursor + N_EXP;               // [T*2]
    float*  topkW  = (float*)(topkId + ROWS);      // [T*2]
    int*    rowTok = (int*)(topkW + ROWS);         // [ROWS]
    float*  rowW   = (float*)(rowTok + ROWS);      // [ROWS]
    __bf16* hbuf   = (__bf16*)(rowW + ROWS);       // [ROWS, I] bf16 (~23 MB)

    // 0) zero output + bookkeeping
    {
        int nthreads = T_TOK * H_DIM;
        k_zero<<<(nthreads + 255) / 256, 256, 0, stream>>>(out, cnt, rowTok, rowW);
    }
    // 1) route
    k_route<<<(T_TOK + 255) / 256, 256, 0, stream>>>(logits, cnt, topkId, topkW);
    // 2) scan
    k_scan<<<1, 32, 0, stream>>>(cnt, off, cursor);
    // 3) scatter
    k_scatter<<<(T_TOK + 255) / 256, 256, 0, stream>>>(topkId, topkW, cursor, rowTok, rowW);
    // 4) GEMM1: silu(xW1^T)*(xW3^T) -> hbuf (bf16)
    {
        dim3 grid(I_DIM / NB, T_TOK / 16, N_EXP);
        k_gemm1<<<grid, 256, 0, stream>>>(X, W1, W3, rowTok, off, cnt, hbuf);
    }
    // 5) GEMM2: hbuf W2^T * gate -> out (atomic scatter-add)
    {
        dim3 grid(H_DIM / NB, T_TOK / 16, N_EXP);
        k_gemm2<<<grid, 256, 0, stream>>>(hbuf, W2, rowTok, rowW, off, cnt, out);
    }
}
